// IGMC_15247133901635
// MI455X (gfx1250) — matrix operand order repaired and checked by reading, not compiled, not run on hardware
//
#include <hip/hip_runtime.h>
#include <hip/hip_bf16.h>
#include <math.h>

// ---------------- problem constants ----------------
#define Gn   5
#define Nn   50000
#define En   250000
#define Bn   50
#define B2   100      // 2*B
#define Ln   200
#define Dn   64
#define Rn   5
#define NBn  2
#define Hn   4
#define LDc  32
#define LPAD 208      // 13*16  (L padded to M-tile multiple)
#define KPAD 224      // 7*32   (L padded to K multiple of 32)

#define DEV __device__ __forceinline__

typedef __attribute__((ext_vector_type(16))) _Float16 v16h;
typedef __attribute__((ext_vector_type(8)))  _Float16 v8h;
typedef __attribute__((ext_vector_type(8)))  float    v8f;

// ---------------- WMMA helpers (CDNA5 16x16x32 f16 -> f32) ----------------
// A fragment: row-major [M x K] f16, tile at (row0, k0).
// ISA 7.12.2 (16-bit A 16x32): lane m=lane&15; lanes<16 carry K kb=0 halves
// {0..7,16..23}, lanes>=16 carry kb=8 halves {8..15,24..31}. Both halves are
// contiguous 16-byte spans -> two ds_load_b128.
DEV v16h load_a(const _Float16* A, int lda, int row0, int k0) {
  int lane = threadIdx.x & 31;
  int m = row0 + (lane & 15);
  int kb = (lane < 16) ? 0 : 8;
  const _Float16* p = A + (size_t)m * lda + k0 + kb;
  v8h lo = *(const v8h*)(p);
  v8h hi = *(const v8h*)(p + 16);
  v16h a;
#pragma unroll
  for (int i = 0; i < 8; ++i) { a[i] = lo[i]; a[i + 8] = hi[i]; }
  return a;
}
// B fragment from a TRANSPOSED buffer Bt (row-major [N x K]): per lane the
// K-values {kb..kb+7, kb+16..kb+23} of row n (kb=0/8) -> two ds_load_b128.
DEV v16h load_bT(const _Float16* Bt, int ldb, int k0, int col0) {
  int lane = threadIdx.x & 31;
  int n = col0 + (lane & 15);
  int kb = (lane < 16) ? 0 : 8;
  const _Float16* p = Bt + (size_t)n * ldb + k0 + kb;
  v8h lo = *(const v8h*)(p);
  v8h hi = *(const v8h*)(p + 16);
  v16h b;
#pragma unroll
  for (int i = 0; i < 8; ++i) { b[i] = lo[i]; b[i + 8] = hi[i]; }
  return b;
}
DEV void store_c_f32(float* C, int ldc, int row0, int col0, v8f c) {
  int lane = threadIdx.x & 31;
  int m0 = row0 + ((lane >> 4) << 3);
  int n = col0 + (lane & 15);
#pragma unroll
  for (int r = 0; r < 8; ++r) C[(size_t)(m0 + r) * ldc + n] = c[r];
}
DEV void store_c_f16(_Float16* C, int ldc, int row0, int col0, v8f c) {
  int lane = threadIdx.x & 31;
  int m0 = row0 + ((lane >> 4) << 3);
  int n = col0 + (lane & 15);
#pragma unroll
  for (int r = 0; r < 8; ++r) C[(size_t)(m0 + r) * ldc + n] = (_Float16)c[r];
}
DEV v8f wmma16(v16h a, v16h b, v8f c) {
  return __builtin_amdgcn_wmma_f32_16x16x32_f16(false, a, false, b, (short)0, c,
                                                false, false);
}

// ---------------- utility kernels ----------------
__global__ void k_zero(float* p, int n) {
  int i = blockIdx.x * 256 + threadIdx.x;
  if (i < n) p[i] = 0.f;
}
// f32 [K x N] -> f16 transposed [N x K] (for contiguous B-fragment loads)
__global__ void k_f16_T(const float* __restrict__ s, _Float16* __restrict__ d,
                        int K, int N) {
  int i = blockIdx.x * 256 + threadIdx.x;
  if (i >= K * N) return;
  int n = i / K, k = i % K;
  d[i] = (_Float16)s[k * N + n];
}
__global__ void k_prep_convW(const float* __restrict__ convW, _Float16* __restrict__ out) {
  int i = blockIdx.x * 256 + threadIdx.x;
  if (i >= LPAD * KPAD) return;
  int o = i / KPAD, l = i % KPAD;
  out[i] = (o < Ln && l < Ln) ? (_Float16)convW[o * Ln + l] : (_Float16)0.f;
}

// ---------------- RGCN kernels ----------------
__global__ void k_cls(const float* __restrict__ xg, int* __restrict__ cls) {
  int n = blockIdx.x * 256 + threadIdx.x;
  if (n >= Nn) return;
  const float* r = xg + (size_t)n * 4;
  int c = 0; float best = r[0];
#pragma unroll
  for (int i = 1; i < 4; ++i) if (r[i] > best) { best = r[i]; c = i; }
  cls[n] = c;
}
__global__ void k_cnt(const int* __restrict__ dst, const int* __restrict__ et,
                      float* __restrict__ cnt) {
  int e = blockIdx.x * 256 + threadIdx.x;
  if (e >= En) return;
  atomicAdd(&cnt[(size_t)dst[e] * Rn + et[e]], 1.0f);
}
// layer 0: x is one-hot -> gather basis rows directly
__global__ void k_xb0(const int* __restrict__ cls, const float* __restrict__ basis0,
                      const float* __restrict__ root0, float* __restrict__ xb,
                      float* __restrict__ rootterm) {
  int t = blockIdx.x * 256 + threadIdx.x;
  if (t >= Nn * 32) return;
  int n = t >> 5, o = t & 31;
  int c = cls[n];
  xb[((size_t)n * 2 + 0) * 32 + o] = basis0[(0 * 4 + c) * 32 + o];
  xb[((size_t)n * 2 + 1) * 32 + o] = basis0[(1 * 4 + c) * 32 + o];
  rootterm[(size_t)n * 32 + o] = root0[c * 32 + o];
}
// recurrent layers: xb = h @ basis[b], rootterm = h @ root
__global__ void k_xbR(const float* __restrict__ h, const float* __restrict__ basis,
                      const float* __restrict__ rootm, float* __restrict__ xb,
                      float* __restrict__ rootterm) {
  int t = blockIdx.x * 256 + threadIdx.x;
  int n = t >> 5, o = t & 31;
  __shared__ float hs[8][32];
  int grp = threadIdx.x >> 5;
  int nb = blockIdx.x * 8 + grp;
  if (nb < Nn) hs[grp][o] = h[(size_t)nb * 32 + o];
  __syncthreads();
  if (n >= Nn) return;
  float a0 = 0.f, a1 = 0.f, rt = 0.f;
#pragma unroll 8
  for (int i = 0; i < 32; ++i) {
    float hv = hs[grp][i];
    a0 = fmaf(hv, basis[(0 * 32 + i) * 32 + o], a0);
    a1 = fmaf(hv, basis[(1 * 32 + i) * 32 + o], a1);
    rt = fmaf(hv, rootm[i * 32 + o], rt);
  }
  xb[((size_t)n * 2 + 0) * 32 + o] = a0;
  xb[((size_t)n * 2 + 1) * 32 + o] = a1;
  rootterm[(size_t)n * 32 + o] = rt;
}
// edge scatter: mean folded in by pre-dividing each message by segment count
__global__ void k_edge(const int* __restrict__ src, const int* __restrict__ dst,
                       const int* __restrict__ et, const float* __restrict__ comp,
                       const float* __restrict__ cnt, const float* __restrict__ xb,
                       float* __restrict__ agg) {
  int t = blockIdx.x * 256 + threadIdx.x;
  if (t >= En * 32) return;
  int e = t >> 5, o = t & 31;
  int s = src[e], d = dst[e], r = et[e];
  float w0 = comp[r * 2 + 0], w1 = comp[r * 2 + 1];
  float c = cnt[(size_t)d * Rn + r];
  c = (c < 1.f) ? 1.f : c;
  float m = (w0 * xb[((size_t)s * 2 + 0) * 32 + o] +
             w1 * xb[((size_t)s * 2 + 1) * 32 + o]) / c;
  atomicAdd(&agg[(size_t)d * 32 + o], m);
}
__global__ void k_node(const float* __restrict__ agg, const float* __restrict__ rootterm,
                       const float* __restrict__ bias, float* __restrict__ h,
                       float* __restrict__ cs, int layer) {
  int t = blockIdx.x * 256 + threadIdx.x;
  if (t >= Nn * 32) return;
  int n = t >> 5, o = t & 31;
  float v = tanhf(agg[(size_t)n * 32 + o] + rootterm[(size_t)n * 32 + o] + bias[o]);
  h[(size_t)n * 32 + o] = v;
  cs[(size_t)n * 128 + layer * 32 + o] = v;
}
// first-B selection of user/item nodes, in index order (single wave scan)
__global__ void k_select(const float* __restrict__ xg, int* __restrict__ usel,
                         int* __restrict__ isel) {
  int lane = threadIdx.x;  // 32 threads
  for (int j = lane; j < Bn; j += 32) { usel[j] = 0; isel[j] = 0; }
  int ucnt = 0, icnt = 0;
  for (int base = 0; base < Nn; base += 32) {
    int n = base + lane;
    bool pu = (n < Nn) && (xg[(size_t)n * 4 + 0] == 1.0f);
    bool pi = (n < Nn) && (xg[(size_t)n * 4 + 1] == 1.0f);
    unsigned long long mu = __ballot(pu);
    unsigned long long mi = __ballot(pi);
    unsigned long long lower = ((unsigned long long)1 << lane) - 1ull;
    int pu_pos = ucnt + __popcll(mu & lower);
    int pi_pos = icnt + __popcll(mi & lower);
    if (pu && pu_pos < Bn) usel[pu_pos] = n;
    if (pi && pi_pos < Bn) isel[pi_pos] = n;
    ucnt += __popcll(mu);
    icnt += __popcll(mi);
    if (ucnt >= Bn && icnt >= Bn) break;
  }
}
__global__ void k_gatherfeat(const float* __restrict__ cs, const int* __restrict__ usel,
                             const int* __restrict__ isel, float* __restrict__ xfeat_g) {
  int t = blockIdx.x * 256 + threadIdx.x;
  if (t >= Bn * 256) return;
  int j = t >> 8, c = t & 255;
  int n = (c < 128) ? usel[j] : isel[j];
  xfeat_g[(size_t)j * 256 + c] = cs[(size_t)n * 128 + (c & 127)];
}

// ---------------- STAM kernels ----------------
// gather emb rows, add pos, project to q/k/v via WMMA (weights pre-transposed)
__global__ void k_stam_proj(const float* __restrict__ emb, const int* __restrict__ ttab,
                            const float* __restrict__ pos, const _Float16* __restrict__ QhT,
                            const _Float16* __restrict__ KhT, const _Float16* __restrict__ VhT,
                            _Float16* __restrict__ t_ws, _Float16* __restrict__ q_ws,
                            _Float16* __restrict__ k_ws, _Float16* __restrict__ v_ws) {
  int b = blockIdx.x;  // 0..99
  __shared__ alignas(16) _Float16 tS[LPAD * Dn];     // 26.6 KB, row-major [L][D]
  __shared__ alignas(16) _Float16 wT[3 * Dn * Dn];   // 24.6 KB, transposed [N][K]
  for (int i = threadIdx.x; i < Dn * Dn; i += 256) {
    wT[i] = QhT[i]; wT[Dn * Dn + i] = KhT[i]; wT[2 * Dn * Dn + i] = VhT[i];
  }
  for (int i = threadIdx.x; i < LPAD * Dn; i += 256) {
    int l = i >> 6, d = i & 63;
    float v = 0.f;
    if (l < Ln) {
      int row = ttab[(size_t)b * Ln + l];
      v = emb[(size_t)row * Dn + d] + pos[l * Dn + d];
      t_ws[((size_t)b * Ln + l) * Dn + d] = (_Float16)v;
    }
    tS[i] = (_Float16)v;
  }
  __syncthreads();
  int wave = threadIdx.x >> 5;
  for (int job = wave; job < 3 * 13 * 4; job += 8) {
    int mat = job / 52, rem = job % 52;
    int mt = rem >> 2, nt = rem & 3;
    v8f c = {};
    for (int k0 = 0; k0 < Dn; k0 += 32) {
      v16h a = load_a(tS, Dn, mt * 16, k0);
      v16h bb = load_bT(wT + mat * Dn * Dn, Dn, k0, nt * 16);
      c = wmma16(a, bb, c);
    }
    _Float16* outp = (mat == 0) ? q_ws : ((mat == 1) ? k_ws : v_ws);
    store_c_f16(outp + (size_t)b * LPAD * Dn, Dn, mt * 16, nt * 16, c);
  }
}

// per-(b,head) attention: QK^T tiles -> masked softmax -> P @ V, all WMMA
__global__ void k_stam_attn(const _Float16* __restrict__ q_ws,
                            const _Float16* __restrict__ k_ws,
                            const _Float16* __restrict__ v_ws,
                            _Float16* __restrict__ h_ws) {
  int b = blockIdx.x >> 2, hd = blockIdx.x & 3;
  __shared__ alignas(16) _Float16 qS[LPAD * 32];   // [l][d], d padded to 32
  __shared__ alignas(16) _Float16 kS[LPAD * 32];   // [m][d] == transposed B
  __shared__ alignas(16) _Float16 vT[16 * KPAD];   // [d][l] transposed V
  __shared__ alignas(16) float    sS[16 * LPAD];
  __shared__ alignas(16) _Float16 pS[16 * KPAD];
  __shared__ float    oPart[4][16 * 16];
  __shared__ float    red[16][8];
  const float scale = 0.0707106781f;  // 1/sqrt(L=200)
  for (int i = threadIdx.x; i < LPAD * 32; i += 128) {
    int l = i >> 5, d = i & 31;
    _Float16 qv = (_Float16)0.f, kv = (_Float16)0.f;
    if (d < 16) {
      qv = q_ws[((size_t)b * LPAD + l) * Dn + hd * 16 + d];
      kv = k_ws[((size_t)b * LPAD + l) * Dn + hd * 16 + d];
    }
    qS[i] = qv; kS[i] = kv;
  }
  for (int i = threadIdx.x; i < 16 * KPAD; i += 128) {
    int d = i / KPAD, l = i % KPAD;
    vT[i] = (l < LPAD) ? v_ws[((size_t)b * LPAD + l) * Dn + hd * 16 + d]
                       : (_Float16)0.f;
  }
  __syncthreads();
  int wave = threadIdx.x >> 5;
  for (int rt = 0; rt < 13; ++rt) {
    for (int ct = wave; ct < 13; ct += 4) {      // S row-tile = q kT
      v16h a = load_a(qS, 32, rt * 16, 0);
      v16h bb = load_bT(kS, 32, 0, ct * 16);
      v8f c = {};
      c = wmma16(a, bb, c);
      store_c_f32(sS, LPAD, 0, ct * 16, c);
    }
    __syncthreads();
    // masked softmax over m<200, 8 threads per row
    int r = threadIdx.x >> 3, sub = threadIdx.x & 7;
    int l = rt * 16 + r;
    float mx = -3.0e38f;
    if (l < Ln) for (int m = sub; m < Ln; m += 8) mx = fmaxf(mx, sS[r * LPAD + m]);
    red[r][sub] = mx;
    __syncthreads();
    mx = red[r][0];
#pragma unroll
    for (int i = 1; i < 8; ++i) mx = fmaxf(mx, red[r][i]);
    mx *= scale;
    float sum = 0.f;
    if (l < Ln) for (int m = sub; m < Ln; m += 8) {
      float e = __expf(sS[r * LPAD + m] * scale - mx);
      sS[r * LPAD + m] = e; sum += e;
    }
    __syncthreads();
    red[r][sub] = sum;
    __syncthreads();
    sum = 0.f;
#pragma unroll
    for (int i = 0; i < 8; ++i) sum += red[r][i];
    float inv = (l < Ln && sum > 0.f) ? 1.f / sum : 0.f;
    for (int m = sub; m < KPAD; m += 8) {
      float pv = (l < Ln && m < Ln) ? sS[r * LPAD + m] * inv : 0.f;
      pS[r * KPAD + m] = (_Float16)pv;
    }
    __syncthreads();
    // out tile = P @ V (K=224 split across 4 waves, then reduced)
    {
      v8f c = {};
      for (int ks = wave; ks < 7; ks += 4) {
        v16h a = load_a(pS, KPAD, 0, ks * 32);
        v16h bb = load_bT(vT, KPAD, ks * 32, 0);
        c = wmma16(a, bb, c);
      }
      store_c_f32(oPart[wave], 16, 0, 0, c);
    }
    __syncthreads();
    for (int i = threadIdx.x; i < 256; i += 128) {
      int r2 = i >> 4, d = i & 15;
      float v = oPart[0][i] + oPart[1][i] + oPart[2][i] + oPart[3][i];
      int l2 = rt * 16 + r2;
      h_ws[((size_t)b * LPAD + l2) * Dn + hd * 16 + d] = (_Float16)v;
    }
    __syncthreads();
  }
}

// k=1 conv over the L axis: out = t + convW@h + convb + h
__global__ void k_stam_conv(const _Float16* __restrict__ convWh, const float* __restrict__ convb,
                            const _Float16* __restrict__ h_ws, const _Float16* __restrict__ t_ws,
                            float* __restrict__ out_ws) {
  int b = blockIdx.x;
  __shared__ alignas(16) _Float16 hT[Dn * KPAD];   // 28.7 KB, transposed [d][l]
  for (int i = threadIdx.x; i < Dn * KPAD; i += 256) {
    int d = i / KPAD, l = i % KPAD;
    hT[i] = (l < LPAD) ? h_ws[((size_t)b * LPAD + l) * Dn + d] : (_Float16)0.f;
  }
  __syncthreads();
  int wave = threadIdx.x >> 5, lane = threadIdx.x & 31;
  for (int job = wave; job < 13 * 4; job += 8) {
    int mt = job >> 2, nt = job & 3;
    v8f c = {};
    for (int k0 = 0; k0 < KPAD; k0 += 32) {
      v16h a = load_a(convWh, KPAD, mt * 16, k0);   // streamed from global
      v16h bb = load_bT(hT, KPAD, k0, nt * 16);
      c = wmma16(a, bb, c);
    }
    int m0 = mt * 16 + ((lane >> 4) << 3);
    int n = nt * 16 + (lane & 15);
#pragma unroll
    for (int rr = 0; rr < 8; ++rr) {
      int o = m0 + rr;
      if (o < Ln) {
        float v = c[rr] + convb[o] + (float)hT[(size_t)n * KPAD + o] +
                  (float)t_ws[((size_t)b * Ln + o) * Dn + n];
        out_ws[((size_t)b * Ln + o) * Dn + n] = v;
      }
    }
  }
}

// row-normalize, temporal weights, masked renorm, weighted emb gather
__global__ void k_stam_final(const float* __restrict__ out_ws, const float* __restrict__ emb,
                             const int* __restrict__ ttab, const int* __restrict__ cidx,
                             const float* __restrict__ tlabel, float* __restrict__ graph_tem,
                             int g) {
  int b = blockIdx.x;
  __shared__ float W1s[Dn];
  __shared__ float sc[Ln];
  __shared__ int   rows[Ln];
  __shared__ float red[256];
  int tid = threadIdx.x;
  if (tid < Dn) W1s[tid] = emb[(size_t)cidx[b] * Dn + tid];
  if (tid < Ln) rows[tid] = ttab[(size_t)b * Ln + tid];
  __syncthreads();
  if (tid < Ln) {
    float dot = 0.f, nn = 0.f;
    const float* orow = out_ws + ((size_t)b * Ln + tid) * Dn;
    for (int d = 0; d < Dn; ++d) { float v = orow[d]; dot = fmaf(v, W1s[d], dot); nn = fmaf(v, v, nn); }
    float nrm = fmaxf(sqrtf(nn), 1e-12f);
    sc[tid] = dot / nrm;
  }
  __syncthreads();
  float v0 = (tid < Ln) ? sc[tid] : -3.0e38f;
  red[tid] = v0; __syncthreads();
  for (int s = 128; s > 0; s >>= 1) { if (tid < s) red[tid] = fmaxf(red[tid], red[tid + s]); __syncthreads(); }
  float mx = red[0]; __syncthreads();
  float e = (tid < Ln) ? __expf(sc[tid] - mx) : 0.f;
  red[tid] = e; __syncthreads();
  for (int s = 128; s > 0; s >>= 1) { if (tid < s) red[tid] += red[tid + s]; __syncthreads(); }
  float ssum = red[0]; __syncthreads();
  float am = (tid < Ln) ? (e / ssum) * tlabel[(size_t)b * Ln + tid] : 0.f;
  red[tid] = am; __syncthreads();
  for (int s = 128; s > 0; s >>= 1) { if (tid < s) red[tid] += red[tid + s]; __syncthreads(); }
  float s2 = red[0]; __syncthreads();
  if (tid < Ln) sc[tid] = (s2 > 0.f) ? am / s2 : 0.f;  // isfinite guard
  __syncthreads();
  if (tid < Dn) {
    float acc = 0.f;
    for (int l = 0; l < Ln; ++l) acc = fmaf(sc[l], emb[(size_t)rows[l] * Dn + tid], acc);
    int i = b >> 1, half = b & 1;
    graph_tem[(size_t)i * (Gn * 2 * Dn) + g * (2 * Dn) + half * Dn + tid] = acc;
  }
}

// ---------------- head kernels ----------------
__global__ void k_att1(const float* __restrict__ xfeat, const float* __restrict__ aw1,
                       const float* __restrict__ ab1, const float* __restrict__ aw2,
                       float* __restrict__ attK) {
  int bid = blockIdx.x; int b = bid / Gn, g = bid % Gn;
  __shared__ float srow[256];
  __shared__ float red[256];
  int j = threadIdx.x;
  srow[j] = xfeat[((size_t)g * Bn + b) * 256 + j];
  __syncthreads();
  float acc = 0.f;
  for (int c = 0; c < 256; ++c) acc = fmaf(srow[c], aw1[(size_t)c * 256 + j], acc);
  acc = fmaxf(acc + ab1[j], 0.f);
  red[j] = acc * aw2[j];
  __syncthreads();
  for (int s = 128; s > 0; s >>= 1) { if (j < s) red[j] += red[j + s]; __syncthreads(); }
  if (j == 0) attK[b * Gn + g] = red[0];
}
__global__ void k_att2(const float* __restrict__ attK, const float* __restrict__ xfeat,
                       float* __restrict__ agg2) {
  int b = blockIdx.x; int c = threadIdx.x;
  float w[Gn]; float mx = -3.0e38f;
#pragma unroll
  for (int g = 0; g < Gn; ++g) { w[g] = attK[b * Gn + g]; mx = fmaxf(mx, w[g]); }
  float s = 0.f;
#pragma unroll
  for (int g = 0; g < Gn; ++g) { w[g] = __expf(w[g] - mx); s += w[g]; }
  float acc = 0.f;
#pragma unroll
  for (int g = 0; g < Gn; ++g) acc = fmaf(w[g] / s, xfeat[((size_t)g * Bn + b) * 256 + c], acc);
  agg2[(size_t)b * 256 + c] = acc;
}
__global__ void k_meta1(const float* __restrict__ agg2, const float* __restrict__ mw1,
                        const float* __restrict__ mb1, float* __restrict__ meta_h) {
  int b = blockIdx.x; int j = threadIdx.x;  // 512 threads
  __shared__ float a[256];
  if (j < 256) a[j] = agg2[(size_t)b * 256 + j];
  __syncthreads();
  float acc = 0.f;
  for (int c = 0; c < 256; ++c) acc = fmaf(a[c], mw1[(size_t)c * 512 + j], acc);
  meta_h[(size_t)b * 512 + j] = fmaxf(acc + mb1[j], 0.f);
}
__global__ void k_meta2(const float* __restrict__ meta_h, const float* __restrict__ mw2,
                        const float* __restrict__ mb2, const float* __restrict__ xfeat,
                        float* __restrict__ xcat1) {
  int b = blockIdx.x;
  __shared__ float h[512];
  for (int i = threadIdx.x; i < 512; i += 256) h[i] = meta_h[(size_t)b * 512 + i];
  __syncthreads();
  for (int j2 = threadIdx.x; j2 < 1280; j2 += 256) {
    float acc = 0.f;
    for (int c = 0; c < 512; ++c) acc = fmaf(h[c], mw2[(size_t)c * 1280 + j2], acc);
    acc += mb2[j2];
    int g = j2 >> 8, c2 = j2 & 255;
    xcat1[(size_t)b * 1280 + j2] = acc * xfeat[((size_t)g * Bn + b) * 256 + c2];
  }
}
__global__ void k_l12(const float* __restrict__ xcat1, const float* __restrict__ gtem,
                      const float* __restrict__ l1w, const float* __restrict__ l1b,
                      const float* __restrict__ l2w, const float* __restrict__ l2b,
                      float* __restrict__ out) {
  int b = blockIdx.x; int k = threadIdx.x;  // 128 threads
  __shared__ float xc[1920];
  __shared__ float red[128];
  for (int i = k; i < 1280; i += 128) xc[i] = xcat1[(size_t)b * 1280 + i];
  for (int i = k; i < 640; i += 128) xc[1280 + i] = gtem[(size_t)b * 640 + i];
  __syncthreads();
  float acc = 0.f;
  for (int c = 0; c < 1920; ++c) acc = fmaf(xc[c], l1w[(size_t)c * 128 + k], acc);
  float hh = fmaxf(acc + l1b[k], 0.f);
  red[k] = hh * l2w[k];
  __syncthreads();
  for (int s = 64; s > 0; s >>= 1) { if (k < s) red[k] += red[k + s]; __syncthreads(); }
  if (k == 0) out[b] = red[0] + l2b[0];
}

// ---------------- launcher ----------------
static inline int cdiv(long a, long b) { return (int)((a + b - 1) / b); }

extern "C" void kernel_launch(void* const* d_in, const int* in_sizes, int n_in,
                              void* d_out, int out_size, void* d_ws, size_t ws_size,
                              hipStream_t stream) {
  (void)in_sizes; (void)n_in; (void)out_size; (void)ws_size;
  const float* x      = (const float*)d_in[0];
  const int*   eidx   = (const int*)d_in[1];
  const int*   etyp   = (const int*)d_in[2];
  const int*   cidx   = (const int*)d_in[3];
  const int*   ttab   = (const int*)d_in[4];
  const float* tlabel = (const float*)d_in[5];
  const float* emb    = (const float*)d_in[6];
  const float* basis0 = (const float*)d_in[7];
  const float* comp0  = (const float*)d_in[8];
  const float* root0  = (const float*)d_in[9];
  const float* bias0  = (const float*)d_in[10];
  const float* basisR = (const float*)d_in[11];
  const float* compR  = (const float*)d_in[12];
  const float* rootR  = (const float*)d_in[13];
  const float* biasR  = (const float*)d_in[14];
  const float* pos    = (const float*)d_in[15];
  const float* Qw     = (const float*)d_in[16];
  const float* Kw     = (const float*)d_in[17];
  const float* Vw     = (const float*)d_in[18];
  const float* convW  = (const float*)d_in[19];
  const float* convb  = (const float*)d_in[20];
  const float* aw1    = (const float*)d_in[21];
  const float* ab1    = (const float*)d_in[22];
  const float* aw2    = (const float*)d_in[23];
  const float* mw1    = (const float*)d_in[24];
  const float* mb1    = (const float*)d_in[25];
  const float* mw2    = (const float*)d_in[26];
  const float* mb2    = (const float*)d_in[27];
  const float* l1w    = (const float*)d_in[28];
  const float* l1b    = (const float*)d_in[29];
  const float* l2w    = (const float*)d_in[30];
  const float* l2b    = (const float*)d_in[31];

  // workspace bump allocator (deterministic layout)
  char* W = (char*)d_ws;
  size_t off = 0;
  auto alloc = [&](size_t bytes) { size_t o = off; off += (bytes + 255) & ~(size_t)255; return o; };
  int*      cls      = (int*)     (W + alloc((size_t)Nn * 4));
  float*    cnt      = (float*)   (W + alloc((size_t)Nn * Rn * 4));
  float*    aggb     = (float*)   (W + alloc((size_t)Nn * 32 * 4));
  float*    xb       = (float*)   (W + alloc((size_t)Nn * 64 * 4));
  float*    rootterm = (float*)   (W + alloc((size_t)Nn * 32 * 4));
  float*    hbuf     = (float*)   (W + alloc((size_t)Nn * 32 * 4));
  float*    cs       = (float*)   (W + alloc((size_t)Nn * 128 * 4));
  int*      usel     = (int*)     (W + alloc((size_t)Bn * 4));
  int*      isel     = (int*)     (W + alloc((size_t)Bn * 4));
  float*    xfeat    = (float*)   (W + alloc((size_t)Gn * Bn * 256 * 4));
  float*    gtem     = (float*)   (W + alloc((size_t)Bn * Gn * 128 * 4));
  _Float16* QhT      = (_Float16*)(W + alloc((size_t)Dn * Dn * 2));
  _Float16* KhT      = (_Float16*)(W + alloc((size_t)Dn * Dn * 2));
  _Float16* VhT      = (_Float16*)(W + alloc((size_t)Dn * Dn * 2));
  _Float16* convWh   = (_Float16*)(W + alloc((size_t)LPAD * KPAD * 2));
  _Float16* t_ws     = (_Float16*)(W + alloc((size_t)B2 * Ln * Dn * 2));
  _Float16* q_ws     = (_Float16*)(W + alloc((size_t)B2 * LPAD * Dn * 2));
  _Float16* k_wsp    = (_Float16*)(W + alloc((size_t)B2 * LPAD * Dn * 2));
  _Float16* v_ws     = (_Float16*)(W + alloc((size_t)B2 * LPAD * Dn * 2));
  _Float16* h_ws     = (_Float16*)(W + alloc((size_t)B2 * LPAD * Dn * 2));
  float*    out_ws   = (float*)   (W + alloc((size_t)B2 * Ln * Dn * 4));
  float*    attK     = (float*)   (W + alloc((size_t)Bn * Gn * 4));
  float*    agg2     = (float*)   (W + alloc((size_t)Bn * 256 * 4));
  float*    meta_hp  = (float*)   (W + alloc((size_t)Bn * 512 * 4));
  float*    xcat1    = (float*)   (W + alloc((size_t)Bn * 1280 * 4));

  // one-time f16 weight prep (transposed for contiguous B-fragment loads)
  k_f16_T<<<cdiv(Dn * Dn, 256), 256, 0, stream>>>(Qw, QhT, Dn, Dn);
  k_f16_T<<<cdiv(Dn * Dn, 256), 256, 0, stream>>>(Kw, KhT, Dn, Dn);
  k_f16_T<<<cdiv(Dn * Dn, 256), 256, 0, stream>>>(Vw, VhT, Dn, Dn);
  k_prep_convW<<<cdiv(LPAD * KPAD, 256), 256, 0, stream>>>(convW, convWh);

  for (int g = 0; g < Gn; ++g) {
    const float* xg   = x + (size_t)g * Nn * 4;
    const int*   srcp = eidx + (size_t)g * 2 * En;
    const int*   dstp = srcp + En;
    const int*   etg  = etyp + (size_t)g * En;

    // ---- RGCN ----
    k_cls<<<cdiv(Nn, 256), 256, 0, stream>>>(xg, cls);
    k_zero<<<cdiv((long)Nn * Rn, 256), 256, 0, stream>>>(cnt, Nn * Rn);
    k_cnt<<<cdiv(En, 256), 256, 0, stream>>>(dstp, etg, cnt);

    // layer 0
    k_xb0<<<cdiv((long)Nn * 32, 256), 256, 0, stream>>>(cls, basis0, root0, xb, rootterm);
    k_zero<<<cdiv((long)Nn * 32, 256), 256, 0, stream>>>(aggb, Nn * 32);
    k_edge<<<cdiv((long)En * 32, 256), 256, 0, stream>>>(srcp, dstp, etg, comp0, cnt, xb, aggb);
    k_node<<<cdiv((long)Nn * 32, 256), 256, 0, stream>>>(aggb, rootterm, bias0, hbuf, cs, 0);
    // layers 1..3
    for (int l = 0; l < 3; ++l) {
      k_xbR<<<cdiv((long)Nn * 32, 256), 256, 0, stream>>>(hbuf, basisR + (size_t)l * 2048,
                                                          rootR + (size_t)l * 1024, xb, rootterm);
      k_zero<<<cdiv((long)Nn * 32, 256), 256, 0, stream>>>(aggb, Nn * 32);
      k_edge<<<cdiv((long)En * 32, 256), 256, 0, stream>>>(srcp, dstp, etg, compR + (size_t)l * Rn * 2,
                                                           cnt, xb, aggb);
      k_node<<<cdiv((long)Nn * 32, 256), 256, 0, stream>>>(aggb, rootterm, biasR + (size_t)l * 32,
                                                           hbuf, cs, l + 1);
    }
    k_select<<<1, 32, 0, stream>>>(xg, usel, isel);
    k_gatherfeat<<<cdiv((long)Bn * 256, 256), 256, 0, stream>>>(cs, usel, isel,
                                                                xfeat + (size_t)g * Bn * 256);

    // ---- STAM ----
    k_stam_proj<<<B2, 256, 0, stream>>>(emb, ttab + (size_t)g * B2 * Ln, pos, QhT, KhT, VhT,
                                        t_ws, q_ws, k_wsp, v_ws);
    k_stam_attn<<<B2 * Hn, 128, 0, stream>>>(q_ws, k_wsp, v_ws, h_ws);
    k_stam_conv<<<B2, 256, 0, stream>>>(convWh, convb, h_ws, t_ws, out_ws);
    k_stam_final<<<B2, 256, 0, stream>>>(out_ws, emb, ttab + (size_t)g * B2 * Ln,
                                         cidx + (size_t)g * B2, tlabel + (size_t)g * B2 * Ln,
                                         gtem, g);
  }

  // ---- head ----
  k_att1<<<Bn * Gn, 256, 0, stream>>>(xfeat, aw1, ab1, aw2, attK);
  k_att2<<<Bn, 256, 0, stream>>>(attK, xfeat, agg2);
  k_meta1<<<Bn, 512, 0, stream>>>(agg2, mw1, mb1, meta_hp);
  k_meta2<<<Bn, 256, 0, stream>>>(meta_hp, mw2, mb2, xfeat, xcat1);
  k_l12<<<Bn, 128, 0, stream>>>(xcat1, gtem, l1w, l1b, l2w, l2b, (float*)d_out);
}
